// FlashAttention_80882824118730
// MI455X (gfx1250) — matrix operand order repaired and checked by reading, not compiled, not run on hardware
//
#include <hip/hip_runtime.h>

// ---------------------------------------------------------------------------
// MI455X (gfx1250) fused attention block.
// fp32 inputs are rounded to bf16; all three GEMM stages + both attention
// matmuls run on v_wmma_f32_16x16x32_bf16 (fp32 accumulation). Compute-bound
// workload (~550 GFLOP vs ~0.5GB HBM traffic), so WMMA throughput dominates.
// GEMMs: 64x64 tile / 4 waves, K-step 64, LDS ping-pong double buffer fed by
// gfx1250 async DMA (global_load_async_to_lds_b128, ASYNCcnt) -> no staging
// VGPRs, no spills, 1 barrier + 1 counter-wait per K-round.
// ---------------------------------------------------------------------------

#define NUM_HEADS 16
#define DIM_HEAD  128
#define SEQ       2048
#define BATCH     4
#define DIM       2048
#define INNER     2048      // NUM_HEADS * DIM_HEAD
#define QKV_N     6144      // 3 * INNER
#define BN        8192      // BATCH * SEQ
#define BH        64        // BATCH * NUM_HEADS
#define VT_STRIDE 72        // padded LDS stride for transposed V

typedef __attribute__((ext_vector_type(16))) __bf16         v16bf;
typedef __attribute__((ext_vector_type(16))) unsigned short v16u;
typedef __attribute__((ext_vector_type(8)))  float          v8f;

__device__ __forceinline__ unsigned short f2bf(float f) {
  union { float f; unsigned int u; } v; v.f = f;
  unsigned int r = v.u + 0x7FFFu + ((v.u >> 16) & 1u);   // round-to-nearest-even
  return (unsigned short)(r >> 16);
}
__device__ __forceinline__ float bf2f(unsigned short s) {
  union { unsigned int u; float f; } v; v.u = ((unsigned int)s) << 16;
  return v.f;
}
__device__ __forceinline__ v16bf as_bf(v16u u) {
  union { v16u u; v16bf b; } c; c.u = u; return c.b;
}
__device__ __forceinline__ v8f wmma_bf16(v16bf a, v16bf b, v8f c) {
  // 8 args: (neg_a, A, neg_b, B, c_mod, C, reuse_a, reuse_b)
  return __builtin_amdgcn_wmma_f32_16x16x32_bf16(false, a, false, b, (short)0, c,
                                                 false, false);
}

// Low 32 bits of a flat pointer to __shared__ = wave-relative LDS byte address.
__device__ __forceinline__ unsigned lds_addr(const void* p) {
  return (unsigned)(unsigned long long)p;
}
// Async DMA: copy 64 contiguous bytes per lane from global to LDS (4 x b128).
// The instruction's inst_offset applies to both the LDS and global addresses.
__device__ __forceinline__ void async_copy64(unsigned lds, const void* g) {
  asm volatile("global_load_async_to_lds_b128 %0, %1, off"
               :: "v"(lds), "v"(g) : "memory");
  asm volatile("global_load_async_to_lds_b128 %0, %1, off offset:16"
               :: "v"(lds), "v"(g) : "memory");
  asm volatile("global_load_async_to_lds_b128 %0, %1, off offset:32"
               :: "v"(lds), "v"(g) : "memory");
  asm volatile("global_load_async_to_lds_b128 %0, %1, off offset:48"
               :: "v"(lds), "v"(g) : "memory");
}
__device__ __forceinline__ void wait_async0() {
  asm volatile("s_wait_asynccnt 0x0" ::: "memory");
}

// A fragment (16x32 bf16). `row` points at this lane's M-row (M = lane & 15).
// Contiguous k order (same as frag_b_row, not the ISA table order): lanes 0-15 hold K=0..15, lanes 16-31 hold K=16..31.
__device__ __forceinline__ v16bf frag_a(const unsigned short* row, int laneHi, int kbase) {
  v16u u;
  const unsigned short* p0 = row + kbase + laneHi * 16;
  const unsigned short* p1 = p0 + 8;
#pragma unroll
  for (int i = 0; i < 8; ++i) { u[i] = p0[i]; u[i + 8] = p1[i]; }
  return as_bf(u);
}
// B fragment (32x16 bf16) gathered from a K-contiguous "B^T" row (N = lane & 15).
// Contiguous k order (same as frag_a, not the ISA table order): lanes 0-15 hold K=0..15, lanes 16-31 hold K=16..31.
__device__ __forceinline__ v16bf frag_b_row(const unsigned short* row, int laneHi, int kbase) {
  v16u u;
  const unsigned short* p = row + kbase + laneHi * 16;
#pragma unroll
  for (int i = 0; i < 16; ++i) u[i] = p[i];
  return as_bf(u);
}

// ---------------------------------------------------------------------------
// Stage 0: fp32 -> bf16 conversion
// ---------------------------------------------------------------------------
__global__ void cvt_f32_bf16(const float* __restrict__ src,
                             unsigned short* __restrict__ dst, int n) {
  int i = blockIdx.x * 256 + threadIdx.x;
  if (i < n) dst[i] = f2bf(src[i]);
}

// ---------------------------------------------------------------------------
// Stage 1: QKV GEMM (M=BN, N=QKV_N, K=DIM), head-major scatter epilogue.
// 128 threads (4 waves), 64x64 tile, K-step 64, async-fed LDS ping-pong.
// ---------------------------------------------------------------------------
__global__ __launch_bounds__(128, 1) void qkv_gemm_scatter(
    const unsigned short* __restrict__ A,    // x  bf16 [BN, DIM]
    const unsigned short* __restrict__ W,    // w_qkv bf16 [QKV_N, DIM] (row-major = B^T)
    unsigned short* __restrict__ Qh,         // [BH, SEQ, DIM_HEAD]
    unsigned short* __restrict__ Kh,
    unsigned short* __restrict__ Vh) {
  __shared__ __align__(16) unsigned short As[2][64 * 64];
  __shared__ __align__(16) unsigned short Bs[2][64 * 64];
  const int n0 = blockIdx.x * 64;
  const int m0 = blockIdx.y * 64;
  const int lane   = threadIdx.x & 31;
  const int wv     = threadIdx.x >> 5;
  const int laneHi = lane >> 4;
  const int ln     = lane & 15;
  const int lrow   = threadIdx.x >> 1;
  const int lcol   = (threadIdx.x & 1) * 32;   // 32 contiguous bf16 per thread

  const unsigned short* gA = A + (size_t)(m0 + lrow) * DIM + lcol;
  const unsigned short* gB = W + (size_t)(n0 + lrow) * DIM + lcol;
  const unsigned ldsA0 = lds_addr(&As[0][lrow * 64 + lcol]);
  const unsigned ldsA1 = lds_addr(&As[1][lrow * 64 + lcol]);
  const unsigned ldsB0 = lds_addr(&Bs[0][lrow * 64 + lcol]);
  const unsigned ldsB1 = lds_addr(&Bs[1][lrow * 64 + lcol]);

  // kick off round 0 tile
  async_copy64(ldsA0, gA);
  async_copy64(ldsB0, gB);

  v8f acc[4];
#pragma unroll
  for (int j = 0; j < 4; ++j)
#pragma unroll
    for (int i = 0; i < 8; ++i) acc[j][i] = 0.0f;

  const int NR = DIM / 64;
  for (int r = 0; r < NR; ++r) {
    const int cur = r & 1;
    wait_async0();       // this round's tile has landed (our lanes)
    __syncthreads();     // all waves' tiles landed; prev round's reads done
    if (r + 1 < NR) {    // DMA next round into the other buffer, overlapped
      const unsigned laN = cur ? ldsA0 : ldsA1;
      const unsigned lbN = cur ? ldsB0 : ldsB1;
      async_copy64(laN, gA + (r + 1) * 64);
      async_copy64(lbN, gB + (r + 1) * 64);
    }
    const unsigned short* Acur = cur ? As[1] : As[0];
    const unsigned short* Bcur = cur ? Bs[1] : Bs[0];
    const unsigned short* arow = Acur + (wv * 16 + ln) * 64;
#pragma unroll
    for (int kk = 0; kk < 2; ++kk) {
      // batch fragment loads into distinct regs, then 4 WMMAs back-to-back
      v16bf a  = frag_a(arow, laneHi, kk * 32);
      v16bf b0 = frag_b_row(Bcur + (0 * 16 + ln) * 64, laneHi, kk * 32);
      v16bf b1 = frag_b_row(Bcur + (1 * 16 + ln) * 64, laneHi, kk * 32);
      v16bf b2 = frag_b_row(Bcur + (2 * 16 + ln) * 64, laneHi, kk * 32);
      v16bf b3 = frag_b_row(Bcur + (3 * 16 + ln) * 64, laneHi, kk * 32);
      acc[0] = wmma_bf16(a, b0, acc[0]);
      acc[1] = wmma_bf16(a, b1, acc[1]);
      acc[2] = wmma_bf16(a, b2, acc[2]);
      acc[3] = wmma_bf16(a, b3, acc[3]);
    }
  }

  // C layout: vgpr r, lane -> M = r + laneHi*8, N = ln. Scatter into Q/K/V heads.
#pragma unroll
  for (int j = 0; j < 4; ++j) {
#pragma unroll
    for (int r = 0; r < 8; ++r) {
      int grow = m0 + wv * 16 + r + laneHi * 8;   // flat (b, n) token index
      int gcol = n0 + j * 16 + ln;                // e index in [0, 3*INNER)
      int bb = grow >> 11, nn = grow & (SEQ - 1);
      int sel = gcol >> 11, rem = gcol & (INNER - 1);
      int hd = rem >> 7, dd = rem & (DIM_HEAD - 1);
      unsigned short* dst = (sel == 0) ? Qh : (sel == 1) ? Kh : Vh;
      dst[(((size_t)bb * NUM_HEADS + hd) * SEQ + nn) * DIM_HEAD + dd] = f2bf(acc[j][r]);
    }
  }
}

// ---------------------------------------------------------------------------
// Stage 2: RoPE in place on Q and K (head-major bf16). One thread per
// (token-row, rotation-pair). sin/cos computed on the fly.
// ---------------------------------------------------------------------------
__global__ void rope_inplace(unsigned short* __restrict__ Qh,
                             unsigned short* __restrict__ Kh) {
  int idx = blockIdx.x * 256 + threadIdx.x;   // BH*SEQ*64 threads exactly
  int i   = idx & 63;                         // pair index in [0, dh/2)
  int row = idx >> 6;                         // bh*SEQ + pos
  int pos = row & (SEQ - 1);
  float inv = __powf(10000.0f, -(float)i / 64.0f);
  float fr  = (float)pos * inv;
  float s, c;
  __sincosf(fr, &s, &c);
  size_t base = (size_t)row * DIM_HEAD;
  float q1 = bf2f(Qh[base + i]), q2 = bf2f(Qh[base + i + 64]);
  Qh[base + i]      = f2bf(q1 * c - q2 * s);
  Qh[base + i + 64] = f2bf(q2 * c + q1 * s);
  float k1 = bf2f(Kh[base + i]), k2 = bf2f(Kh[base + i + 64]);
  Kh[base + i]      = f2bf(k1 * c - k2 * s);
  Kh[base + i + 64] = f2bf(k2 * c + k1 * s);
}

// ---------------------------------------------------------------------------
// Stage 3: flash attention. One block per (b, h, 64-query tile); 128 threads.
// QK^T and PV via WMMA bf16; V kept transposed in LDS; online softmax in LDS.
// ---------------------------------------------------------------------------
__global__ __launch_bounds__(128, 1) void attn_kernel(
    const unsigned short* __restrict__ Qh,
    const unsigned short* __restrict__ Kh,
    const unsigned short* __restrict__ Vh,
    unsigned short* __restrict__ Oc) {     // [b, n, h*dh] bf16
  __shared__ __align__(16) unsigned short Qs[64 * DIM_HEAD];
  __shared__ __align__(16) unsigned short Ks[64 * DIM_HEAD];
  __shared__ __align__(16) unsigned short Vt[DIM_HEAD * VT_STRIDE]; // [dh][key]
  __shared__ __align__(16) unsigned short Ss[64 * 64];     // S, then P (bf16)
  __shared__ float m_i[64], l_i[64], p_scale[64];

  const int bh = blockIdx.x >> 5;          // 32 query tiles per head
  const int mt = blockIdx.x & 31;
  const int m0 = mt * 64;
  const int b  = bh >> 4;
  const int h  = bh & (NUM_HEADS - 1);
  const unsigned short* Qp = Qh + (size_t)bh * SEQ * DIM_HEAD;
  const unsigned short* Kp = Kh + (size_t)bh * SEQ * DIM_HEAD;
  const unsigned short* Vp = Vh + (size_t)bh * SEQ * DIM_HEAD;

  const int lane   = threadIdx.x & 31;
  const int wv     = threadIdx.x >> 5;
  const int laneHi = lane >> 4;
  const int ln     = lane & 15;
  const int lrow   = threadIdx.x >> 1;
  const int lhalf  = threadIdx.x & 1;
  const float sm_scale = 0.08838834764831845f;   // 1/sqrt(128)

  // load Q tile (64 x 128 bf16): each thread copies 64 contiguous bf16.
  {
    const uint4* g = (const uint4*)(Qp + (size_t)(m0 + lrow) * DIM_HEAD + lhalf * 64);
    uint4* l = (uint4*)(Qs + lrow * DIM_HEAD + lhalf * 64);
#pragma unroll
    for (int i = 0; i < 8; ++i) l[i] = g[i];
  }
  if (threadIdx.x < 64) { m_i[threadIdx.x] = -1e30f; l_i[threadIdx.x] = 0.0f; }

  v8f o[8];
#pragma unroll
  for (int nt = 0; nt < 8; ++nt)
#pragma unroll
    for (int i = 0; i < 8; ++i) o[nt][i] = 0.0f;
  __syncthreads();

  for (int kt = 0; kt < SEQ / 64; ++kt) {
    const int k0 = kt * 64;
    {
      // K rows straight in; V transposed (dh-major) for contiguous B-fragments.
      const uint4* gk = (const uint4*)(Kp + (size_t)(k0 + lrow) * DIM_HEAD + lhalf * 64);
      uint4* lk = (uint4*)(Ks + lrow * DIM_HEAD + lhalf * 64);
      const uint4* gv = (const uint4*)(Vp + (size_t)(k0 + lrow) * DIM_HEAD + lhalf * 64);
#pragma unroll
      for (int i = 0; i < 8; ++i) {
        lk[i] = gk[i];
        uint4 vv = gv[i];
        const unsigned short* pv = (const unsigned short*)&vv;
#pragma unroll
        for (int e = 0; e < 8; ++e) {
          int dh = lhalf * 64 + i * 8 + e;
          Vt[dh * VT_STRIDE + lrow] = pv[e];
        }
      }
    }
    // gfx1250 prefetch of next tile (global_prefetch_b8) to hide HBM latency.
    if (kt + 1 < SEQ / 64) {
      __builtin_prefetch(Kp + (size_t)(k0 + 64 + lrow) * DIM_HEAD + lhalf * 64, 0, 1);
      __builtin_prefetch(Vp + (size_t)(k0 + 64 + lrow) * DIM_HEAD + lhalf * 64, 0, 1);
    }
    __syncthreads();

    // S = Q K^T : wave handles 16 query rows x 64 keys = 4 C fragments.
    v8f s[4];
#pragma unroll
    for (int j = 0; j < 4; ++j)
#pragma unroll
      for (int i = 0; i < 8; ++i) s[j][i] = 0.0f;
#pragma unroll
    for (int kkc = 0; kkc < DIM_HEAD / 32; ++kkc) {
      v16bf a  = frag_a(Qs + (wv * 16 + ln) * DIM_HEAD, laneHi, kkc * 32);
      v16bf b0 = frag_b_row(Ks + (0 * 16 + ln) * DIM_HEAD, laneHi, kkc * 32);
      v16bf b1 = frag_b_row(Ks + (1 * 16 + ln) * DIM_HEAD, laneHi, kkc * 32);
      v16bf b2 = frag_b_row(Ks + (2 * 16 + ln) * DIM_HEAD, laneHi, kkc * 32);
      v16bf b3 = frag_b_row(Ks + (3 * 16 + ln) * DIM_HEAD, laneHi, kkc * 32);
      s[0] = wmma_bf16(a, b0, s[0]);
      s[1] = wmma_bf16(a, b1, s[1]);
      s[2] = wmma_bf16(a, b2, s[2]);
      s[3] = wmma_bf16(a, b3, s[3]);
    }
    // spill pre-softmax scores (scaled) as bf16
#pragma unroll
    for (int j = 0; j < 4; ++j)
#pragma unroll
      for (int r = 0; r < 8; ++r)
        Ss[(wv * 16 + r + laneHi * 8) * 64 + j * 16 + ln] = f2bf(s[j][r] * sm_scale);
    __syncthreads();

    // online softmax: one thread per query row
    if (threadIdx.x < 64) {
      const int rr = threadIdx.x;
      float mx = -1e30f;
#pragma unroll 8
      for (int cc = 0; cc < 64; ++cc) mx = fmaxf(mx, bf2f(Ss[rr * 64 + cc]));
      float newm = fmaxf(m_i[rr], mx);
      float rsum = 0.0f;
#pragma unroll 8
      for (int cc = 0; cc < 64; ++cc) {
        float p = __expf(bf2f(Ss[rr * 64 + cc]) - newm);
        Ss[rr * 64 + cc] = f2bf(p);
        rsum += p;
      }
      float sc = __expf(m_i[rr] - newm);
      l_i[rr] = l_i[rr] * sc + rsum;
      m_i[rr] = newm;
      p_scale[rr] = sc;
    }
    __syncthreads();

    // rescale O accumulators by exp(old_max - new_max)
#pragma unroll
    for (int r = 0; r < 8; ++r) {
      float sc = p_scale[wv * 16 + r + laneHi * 8];
#pragma unroll
      for (int nt = 0; nt < 8; ++nt) o[nt][r] *= sc;
    }

    // O += P V : P (16x64) as A over two 32-deep steps; V^T rows as B.
#pragma unroll
    for (int kk2 = 0; kk2 < 2; ++kk2) {
      v16bf a = frag_a(Ss + (wv * 16 + ln) * 64, laneHi, kk2 * 32);
#pragma unroll
      for (int g = 0; g < 2; ++g) {       // two groups of 4 batched fragments
        v16bf c0 = frag_b_row(Vt + ((g * 4 + 0) * 16 + ln) * VT_STRIDE, laneHi, kk2 * 32);
        v16bf c1 = frag_b_row(Vt + ((g * 4 + 1) * 16 + ln) * VT_STRIDE, laneHi, kk2 * 32);
        v16bf c2 = frag_b_row(Vt + ((g * 4 + 2) * 16 + ln) * VT_STRIDE, laneHi, kk2 * 32);
        v16bf c3 = frag_b_row(Vt + ((g * 4 + 3) * 16 + ln) * VT_STRIDE, laneHi, kk2 * 32);
        o[g * 4 + 0] = wmma_bf16(a, c0, o[g * 4 + 0]);
        o[g * 4 + 1] = wmma_bf16(a, c1, o[g * 4 + 1]);
        o[g * 4 + 2] = wmma_bf16(a, c2, o[g * 4 + 2]);
        o[g * 4 + 3] = wmma_bf16(a, c3, o[g * 4 + 3]);
      }
    }
    __syncthreads();   // protect Ks/Vt/Ss before next tile load
  }

  // normalize and store to [b, n, h*dh] bf16 for the output projection
#pragma unroll
  for (int r = 0; r < 8; ++r) {
    int M = wv * 16 + r + laneHi * 8;
    float invl = 1.0f / l_i[M];
    int n = m0 + M;
#pragma unroll
    for (int nt = 0; nt < 8; ++nt) {
      Oc[((size_t)b * SEQ + n) * INNER + h * DIM_HEAD + nt * 16 + ln] =
          f2bf(o[nt][r] * invl);
    }
  }
}

// ---------------------------------------------------------------------------
// Stage 4: output GEMM (M=BN, N=DIM, K=INNER) + bias, fp32 result.
// Same async-fed ping-pong structure as stage 1.
// ---------------------------------------------------------------------------
__global__ __launch_bounds__(128, 1) void out_gemm(
    const unsigned short* __restrict__ A,    // Oc bf16 [BN, INNER]
    const unsigned short* __restrict__ W,    // w_out bf16 [DIM, INNER] (row-major = B^T)
    const float* __restrict__ bias,
    float* __restrict__ out) {
  __shared__ __align__(16) unsigned short As[2][64 * 64];
  __shared__ __align__(16) unsigned short Bs[2][64 * 64];
  const int n0 = blockIdx.x * 64;
  const int m0 = blockIdx.y * 64;
  const int lane   = threadIdx.x & 31;
  const int wv     = threadIdx.x >> 5;
  const int laneHi = lane >> 4;
  const int ln     = lane & 15;
  const int lrow   = threadIdx.x >> 1;
  const int lcol   = (threadIdx.x & 1) * 32;

  const unsigned short* gA = A + (size_t)(m0 + lrow) * INNER + lcol;
  const unsigned short* gB = W + (size_t)(n0 + lrow) * INNER + lcol;
  const unsigned ldsA0 = lds_addr(&As[0][lrow * 64 + lcol]);
  const unsigned ldsA1 = lds_addr(&As[1][lrow * 64 + lcol]);
  const unsigned ldsB0 = lds_addr(&Bs[0][lrow * 64 + lcol]);
  const unsigned ldsB1 = lds_addr(&Bs[1][lrow * 64 + lcol]);

  async_copy64(ldsA0, gA);
  async_copy64(ldsB0, gB);

  v8f acc[4];
#pragma unroll
  for (int j = 0; j < 4; ++j)
#pragma unroll
    for (int i = 0; i < 8; ++i) acc[j][i] = 0.0f;

  const int NR = INNER / 64;
  for (int r = 0; r < NR; ++r) {
    const int cur = r & 1;
    wait_async0();
    __syncthreads();
    if (r + 1 < NR) {
      const unsigned laN = cur ? ldsA0 : ldsA1;
      const unsigned lbN = cur ? ldsB0 : ldsB1;
      async_copy64(laN, gA + (r + 1) * 64);
      async_copy64(lbN, gB + (r + 1) * 64);
    }
    const unsigned short* Acur = cur ? As[1] : As[0];
    const unsigned short* Bcur = cur ? Bs[1] : Bs[0];
    const unsigned short* arow = Acur + (wv * 16 + ln) * 64;
#pragma unroll
    for (int kk = 0; kk < 2; ++kk) {
      v16bf a  = frag_a(arow, laneHi, kk * 32);
      v16bf b0 = frag_b_row(Bcur + (0 * 16 + ln) * 64, laneHi, kk * 32);
      v16bf b1 = frag_b_row(Bcur + (1 * 16 + ln) * 64, laneHi, kk * 32);
      v16bf b2 = frag_b_row(Bcur + (2 * 16 + ln) * 64, laneHi, kk * 32);
      v16bf b3 = frag_b_row(Bcur + (3 * 16 + ln) * 64, laneHi, kk * 32);
      acc[0] = wmma_bf16(a, b0, acc[0]);
      acc[1] = wmma_bf16(a, b1, acc[1]);
      acc[2] = wmma_bf16(a, b2, acc[2]);
      acc[3] = wmma_bf16(a, b3, acc[3]);
    }
  }
#pragma unroll
  for (int j = 0; j < 4; ++j) {
#pragma unroll
    for (int r = 0; r < 8; ++r) {
      int grow = m0 + wv * 16 + r + laneHi * 8;
      int gcol = n0 + j * 16 + ln;
      out[(size_t)grow * DIM + gcol] = acc[j][r] + bias[gcol];
    }
  }
}

// ---------------------------------------------------------------------------
// Host launch
// ---------------------------------------------------------------------------
extern "C" void kernel_launch(void* const* d_in, const int* in_sizes, int n_in,
                              void* d_out, int out_size, void* d_ws, size_t ws_size,
                              hipStream_t stream) {
  const float* x     = (const float*)d_in[0];   // [4, 2048, 2048]
  const float* w_qkv = (const float*)d_in[1];   // [6144, 2048]
  const float* w_out = (const float*)d_in[2];   // [2048, 2048]
  const float* b_out = (const float*)d_in[3];   // [2048]
  float* out = (float*)d_out;

  // workspace layout (bf16 buffers as ushort)
  unsigned short* xb  = (unsigned short*)d_ws;                 // BN*DIM
  unsigned short* wqb = xb  + (size_t)BN * DIM;                // QKV_N*DIM
  unsigned short* wob = wqb + (size_t)QKV_N * DIM;             // DIM*INNER
  unsigned short* Qh  = wob + (size_t)DIM * INNER;             // BH*SEQ*DIM_HEAD
  unsigned short* Kh  = Qh  + (size_t)BH * SEQ * DIM_HEAD;
  unsigned short* Vh  = Kh  + (size_t)BH * SEQ * DIM_HEAD;
  unsigned short* Oc  = Vh  + (size_t)BH * SEQ * DIM_HEAD;     // BN*INNER

  const int n_x  = BN * DIM;
  const int n_wq = QKV_N * DIM;
  const int n_wo = DIM * INNER;
  cvt_f32_bf16<<<(n_x  + 255) / 256, 256, 0, stream>>>(x,     xb,  n_x);
  cvt_f32_bf16<<<(n_wq + 255) / 256, 256, 0, stream>>>(w_qkv, wqb, n_wq);
  cvt_f32_bf16<<<(n_wo + 255) / 256, 256, 0, stream>>>(w_out, wob, n_wo);

  qkv_gemm_scatter<<<dim3(QKV_N / 64, BN / 64), 128, 0, stream>>>(xb, wqb, Qh, Kh, Vh);

  rope_inplace<<<(BH * SEQ * 64) / 256, 256, 0, stream>>>(Qh, Kh);

  attn_kernel<<<BH * (SEQ / 64), 128, 0, stream>>>(Qh, Kh, Vh, Oc);

  out_gemm<<<dim3(DIM / 64, BN / 64), 128, 0, stream>>>(Oc, wob, b_out, out);
}
